// Local_based_attention_variation_29463475650711
// MI455X (gfx1250) — matrix operand order repaired and checked by reading, not compiled, not run on hardware
//
#include <hip/hip_runtime.h>
#include <hip/hip_bf16.h>
#include <math.h>

typedef __attribute__((ext_vector_type(16))) _Float16 v16h;
typedef __attribute__((ext_vector_type(8)))  float    v8f;

#define B_ 2
#define H_ 8
#define N_ 4096
#define K_ 32
#define D_ 32
#define WAVES_PER_BLOCK 8

// Native V_TANH_F32 (confirmed present on gfx1250 toolchain); branch-free
// fallback kept for portability.
#if __has_builtin(__builtin_amdgcn_tanh_f32)
__device__ __forceinline__ float fast_tanh(float x) { return __builtin_amdgcn_tanh_f32(x); }
#elif __has_builtin(__builtin_amdgcn_tanhf)
__device__ __forceinline__ float fast_tanh(float x) { return __builtin_amdgcn_tanhf(x); }
#else
__device__ __forceinline__ float fast_tanh(float x) {
    const float ax = __builtin_fabsf(x);
    const float e  = __expf(2.0f * ax);
    const float r  = __builtin_amdgcn_rcpf(e + 1.0f);
    const float t  = __builtin_fmaf(-2.0f, r, 1.0f);
    return __builtin_copysignf(t, x);
}
#endif

__device__ __forceinline__ float bcast_lane(float v, int srcLane) {
    return __int_as_float(__builtin_amdgcn_readlane(__float_as_int(v), srcLane));
}

__global__ __launch_bounds__(WAVES_PER_BLOCK * 32)
void local_attn_wmma_kernel(const float* __restrict__ q,
                            const float* __restrict__ kmat,
                            const float* __restrict__ v,
                            const float* __restrict__ p_add,
                            float* __restrict__ out)
{
    const int tid  = threadIdx.x;
    const int lane = tid & 31;
    const int wv   = tid >> 5;
    const long p   = (long)blockIdx.x * WAVES_PER_BLOCK + wv;   // 0 .. B*H*N-1

    const int n  = (int)(p & (N_ - 1));   // p % 4096
    const int bh = (int)(p >> 12);        // p / 4096
    const int h  = bh & (H_ - 1);
    const int b  = bh >> 3;

    const float* qp = q    + p * D_;            // (..,1,D)
    const float* kp = kmat + p * (D_ * K_);     // (..,D,K) row-major
    const float* vp = v    + p * (K_ * D_);     // (..,K,D) row-major
    float*       op = out  + (((long)b * N_ + n) * H_ + h) * D_;  // (B,N,H,D)

    // Prefetch this position's v tile (each lane prefetches one 128B row start)
    __builtin_prefetch(vp + lane * D_, 0, 1);

    const int g = lane >> 4;         // lane half
    const int m = lane & 15;         // row within fragment

    // ---- q / p_add pairs loaded straight from global (uniform per lane-half,
    // one 128B line each -> L1 broadcast, no shuffles).
    // A-fragment contraction pattern: c = {8g+2j | j<4} U {16+8g+2(j-4) | j>=4}
    float2 qq[8];
#pragma unroll
    for (int j = 0; j < 8; ++j) {
        const int c0 = ((j < 4) ? (2 * j) : (16 + 2 * (j - 4))) + 8 * g;
        qq[j] = *(const float2*)(qp + c0);
    }

    // B fragment (32x16 f16): B[c][n] = p_add[c]; B layout: lane half g holds
    // contraction rows c = {8g+2j | j<4} U {16+8g+2(j-4) | j>=4} (+0, +1).
    v16h bfrag;
#pragma unroll
    for (int j = 0; j < 8; ++j) {
        const float2 pb = *(const float2*)(p_add + ((j < 4) ? (2 * j) : (16 + 2 * (j - 4))) + 8 * g);
        bfrag[2 * j]     = (_Float16)pb.x;
        bfrag[2 * j + 1] = (_Float16)pb.y;
    }

    v8f acc0 = {0.f, 0.f, 0.f, 0.f, 0.f, 0.f, 0.f, 0.f};
    v8f acc1 = acc0;

    // ---- Two WMMAs: scores[k] = sum_d tanh(q[d] + kmat[d][k]) * p_add[d]
#pragma unroll
    for (int half = 0; half < 2; ++half) {
        const int kidx = half * 16 + m;          // neighbor index = A row
        v16h afrag;
#pragma unroll
        for (int j = 0; j < 8; ++j) {
            const int c0 = ((j < 4) ? (2 * j) : (16 + 2 * (j - 4))) + 8 * g;
            afrag[2 * j]     = (_Float16)fast_tanh(qq[j].x + kp[c0       * K_ + kidx]);
            afrag[2 * j + 1] = (_Float16)fast_tanh(qq[j].y + kp[(c0 + 1) * K_ + kidx]);
        }
        if (half == 0)
            acc0 = __builtin_amdgcn_wmma_f32_16x16x32_f16(false, afrag, false, bfrag,
                                                          (short)0, acc0, false, false);
        else
            acc1 = __builtin_amdgcn_wmma_f32_16x16x32_f16(false, afrag, false, bfrag,
                                                          (short)0, acc1, false, false);
    }

    // C/D layout: acc0[r] = scores[8g + r], acc1[r] = scores[16 + 8g + r].
    // Partials are identical across all 16 lanes of a half, so the cross-half
    // combine is just two v_readlane broadcasts (lanes 0 and 16) - no LDS.
    float mx = -INFINITY;
#pragma unroll
    for (int r = 0; r < 8; ++r) mx = fmaxf(mx, fmaxf(acc0[r], acc1[r]));
    mx = fmaxf(bcast_lane(mx, 0), bcast_lane(mx, 16));

    float e0[8], e1[8];
    float sum = 0.f;
#pragma unroll
    for (int r = 0; r < 8; ++r) {
        e0[r] = __expf(acc0[r] - mx);
        e1[r] = __expf(acc1[r] - mx);
        sum += e0[r] + e1[r];
    }
    sum = bcast_lane(sum, 0) + bcast_lane(sum, 16);
    const float inv = 1.0f / sum;

    // x[d] = sum_k attn[k] * v[k][d], d = lane; coalesced 128B v-row loads.
    // Broadcasts use v_readlane (uniform source lanes 0 / 16).
    float x = 0.f;
#pragma unroll
    for (int r = 0; r < 8; ++r) {
        const float w0 = bcast_lane(e0[r], 0);    // k = r       (lane 0:  g=0)
        const float w1 = bcast_lane(e0[r], 16);   // k = 8 + r   (lane 16: g=1)
        const float w2 = bcast_lane(e1[r], 0);    // k = 16 + r
        const float w3 = bcast_lane(e1[r], 16);   // k = 24 + r
        x = fmaf(w0, vp[(r     ) * D_ + lane], x);
        x = fmaf(w1, vp[(r +  8) * D_ + lane], x);
        x = fmaf(w2, vp[(r + 16) * D_ + lane], x);
        x = fmaf(w3, vp[(r + 24) * D_ + lane], x);
    }
    op[lane] = x * inv;
}

extern "C" void kernel_launch(void* const* d_in, const int* in_sizes, int n_in,
                              void* d_out, int out_size, void* d_ws, size_t ws_size,
                              hipStream_t stream) {
    const float* q  = (const float*)d_in[0];
    const float* k  = (const float*)d_in[1];
    const float* v  = (const float*)d_in[2];
    const float* pa = (const float*)d_in[3];
    float* out = (float*)d_out;

    const int positions = B_ * H_ * N_;                 // 65536 wave-sized tasks
    dim3 grid(positions / WAVES_PER_BLOCK);             // 8192 blocks
    dim3 block(WAVES_PER_BLOCK * 32);                   // 8 wave32s per block

    hipLaunchKernelGGL(local_attn_wmma_kernel, grid, block, 0, stream,
                       q, k, v, pa, out);
}
